// Model_32418413150239
// MI455X (gfx1250) — matrix operand order repaired and checked by reading, not compiled, not run on hardware
//
#include <hip/hip_runtime.h>
#include <cstdint>

#ifndef USE_TDM
#define USE_TDM 1
#endif

typedef __attribute__((ext_vector_type(16))) __bf16       v16bf;
typedef __attribute__((ext_vector_type(8)))  float        v8f;
typedef __attribute__((ext_vector_type(4))) unsigned int  u32x4;
typedef __attribute__((ext_vector_type(8)))  int          i32x8;
typedef __attribute__((ext_vector_type(4)))  int          i32x4;

#define BM 128
#define BN 128
#define BK 32
#define LDA 40   // 32 bf16 + 8 pad = 80 B row stride (matches TDM pad 16B/64B)

__device__ __forceinline__ uint16_t f32_to_bf16(float f) {
  uint32_t u = __float_as_uint(f);
  uint32_t r = u + 0x7FFFu + ((u >> 16) & 1u);   // round-to-nearest-even
  return (uint16_t)(r >> 16);
}

union FragU { uint4 q[2]; v16bf v; };

__device__ __forceinline__ uint32_t lds_offset(const void* p) {
  // flat LDS address: low 32 bits are the LDS byte offset
  return (uint32_t)(uintptr_t)p;
}

#if USE_TDM
// TDM 2-D tile load: 128 rows x 32 bf16 (64B/row) from global -> LDS,
// with 16B padding after every 64B row (LDS row stride = 80B = LDA*2).
__device__ __forceinline__ void tdm_load_tile(const uint16_t* gptr,
                                              uint32_t lds_byte_off,
                                              uint32_t row_stride_elems) {
  const uint64_t ga = (uint64_t)(uintptr_t)gptr;
  u32x4 g0;
  g0[0] = 1u;                                    // count=1 (valid), user mode, no gather
  g0[1] = lds_byte_off;                          // lds_addr (bytes)
  g0[2] = (uint32_t)ga;                          // global_addr[31:0]
  g0[3] = (uint32_t)((ga >> 32) & 0x01FFFFFFull) // global_addr[56:32]
        | (2u << 30);                            // type = 2 ("image")
  i32x8 g1;
  g1[0] = (int)((1u << 16)                       // data_size = 2 bytes
              | (1u << 20)                       // pad_enable
              | (3u << 22)                       // pad_interval: 16 dwords (64B row)
              | (3u << 25));                     // pad_amount : 4 dwords  (16B pad)
  g1[1] = (int)((row_stride_elems & 0xFFFFu) << 16);          // tensor_dim0[15:0]
  g1[2] = (int)(((row_stride_elems >> 16) & 0xFFFFu)          // tensor_dim0[31:16]
              | (128u << 16));                                // tensor_dim1[15:0]=128
  g1[3] = (int)(32u << 16);                                   // tensor_dim1 hi=0 | tile_dim0=32
  g1[4] = (int)128u;                                          // tile_dim1=128, tile_dim2=0
  g1[5] = (int)row_stride_elems;                              // tensor_dim0_stride[31:0]
  g1[6] = 0;                                                  // stride0 hi16 | dim1_stride lo16
  g1[7] = 0;
  const i32x4 gz4 = {0, 0, 0, 0};                             // groups 2/3: zero (2-D tensor)
  const i32x8 gz8 = {0, 0, 0, 0, 0, 0, 0, 0};                 // trailing group: zero
  __builtin_amdgcn_tensor_load_to_lds(g0, g1, gz4, gz4, gz8, 0);
}
#endif

// C[E,M,N](bf16 or f32) = A[E,M,K](bf16, row-major) * Bt[E,N,K](bf16, N-major)
// grid: (N/128, M/128, E), block: 256 threads (8 waves of 32)
template <bool OUT_BF16>
__global__ __launch_bounds__(256) void gemm_bf16_wmma(
    const uint16_t* __restrict__ A, const uint16_t* __restrict__ Bt,
    void* __restrict__ Cout, int M, int N, int K) {
  __shared__ uint16_t As[2][BM * LDA];
  __shared__ uint16_t Bs[2][BN * LDA];

  const int e  = blockIdx.z;
  const int m0 = blockIdx.y * BM;
  const int n0 = blockIdx.x * BN;
  const uint16_t* Ae = A  + ((size_t)e * M + m0) * K;
  const uint16_t* Be = Bt + ((size_t)e * N + n0) * K;

  const int lane = threadIdx.x & 31;
  const int wave = threadIdx.x >> 5;
  const int wm = wave >> 2;      // 0..1 : 64 output rows per wave
  const int wn = wave & 3;       // 0..3 : 32 output cols per wave
  const int r  = lane & 15;
  const int hs = lane >> 4;

  v8f acc[4][2];
  #pragma unroll
  for (int i = 0; i < 4; ++i)
    #pragma unroll
    for (int j = 0; j < 2; ++j) acc[i][j] = {};

  const int KT = K / BK;

#if USE_TDM
  if (threadIdx.x < 32) {        // wave 0 drives the TDM pipeline
    tdm_load_tile(Ae, lds_offset(&As[0][0]), (uint32_t)K);
    tdm_load_tile(Be, lds_offset(&Bs[0][0]), (uint32_t)K);
  }
#endif

  for (int kt = 0; kt < KT; ++kt) {
    const int buf = kt & 1;
#if USE_TDM
    if (threadIdx.x < 32) {
      if (kt + 1 < KT) {
        tdm_load_tile(Ae + (size_t)(kt + 1) * BK, lds_offset(&As[buf ^ 1][0]), (uint32_t)K);
        tdm_load_tile(Be + (size_t)(kt + 1) * BK, lds_offset(&Bs[buf ^ 1][0]), (uint32_t)K);
        __builtin_amdgcn_s_wait_tensorcnt(2);  // oldest 2 (current buf) done
      } else {
        __builtin_amdgcn_s_wait_tensorcnt(0);
      }
    }
    __syncthreads();
#else
    __syncthreads();
    #pragma unroll
    for (int it = 0; it < 2; ++it) {           // 512 x 16B chunks / 256 threads
      const int c   = (int)threadIdx.x + it * 256;
      const int row = c >> 2, ch = c & 3;
      *(uint4*)&As[buf][row * LDA + ch * 8] =
          *(const uint4*)(Ae + (size_t)row * K + (size_t)kt * BK + ch * 8);
      *(uint4*)&Bs[buf][row * LDA + ch * 8] =
          *(const uint4*)(Be + (size_t)row * K + (size_t)kt * BK + ch * 8);
    }
    __syncthreads();
#endif

    // A fragment (16x32 bf16): lane hs=0 -> K{0..7,16..23}, hs=1 -> K{8..15,24..31}
    FragU fa[4], fb[2];
    const uint16_t* arow = &As[buf][(wm * 64 + r) * LDA];
    #pragma unroll
    for (int mt = 0; mt < 4; ++mt) {
      const uint16_t* p = arow + mt * 16 * LDA;
      fa[mt].q[0] = *(const uint4*)(p + hs * 8);
      fa[mt].q[1] = *(const uint4*)(p + 16 + hs * 8);
    }
    // B fragment (32x16 bf16): lane = column, hs=0 -> K{0..7,16..23}, hs=1 -> K{8..15,24..31}
    const uint16_t* brow = &Bs[buf][(wn * 32 + r) * LDA];
    #pragma unroll
    for (int nt = 0; nt < 2; ++nt) {
      const uint16_t* p = brow + nt * 16 * LDA;
      fb[nt].q[0] = *(const uint4*)(p + hs * 8);
      fb[nt].q[1] = *(const uint4*)(p + 16 + hs * 8);
    }
    #pragma unroll
    for (int mt = 0; mt < 4; ++mt)
      #pragma unroll
      for (int nt = 0; nt < 2; ++nt)
        acc[mt][nt] = __builtin_amdgcn_wmma_f32_16x16x32_bf16(
            false, fa[mt].v, false, fb[nt].v, (short)0, acc[mt][nt], false, false);

#if USE_TDM
    __syncthreads();   // everyone done reading buf before it is re-filled
#endif
  }

  // C layout: VGPR i -> M = i + 8*hs, N = lane&15
  const size_t cbase = (size_t)e * M * N + (size_t)(m0 + wm * 64) * N + (n0 + wn * 32);
  #pragma unroll
  for (int mt = 0; mt < 4; ++mt)
    #pragma unroll
    for (int nt = 0; nt < 2; ++nt) {
      const int col = nt * 16 + r;
      #pragma unroll
      for (int i = 0; i < 8; ++i) {
        const int row = mt * 16 + hs * 8 + i;
        const size_t idx = cbase + (size_t)row * N + col;
        if (OUT_BF16) ((uint16_t*)Cout)[idx] = f32_to_bf16(acc[mt][nt][i]);
        else          ((float*)Cout)[idx]    = acc[mt][nt][i];
      }
    }
}

__global__ __launch_bounds__(256) void cvt_f32_to_bf16_kernel(
    const float* __restrict__ in, uint16_t* __restrict__ out, size_t n) {
  const size_t i = ((size_t)blockIdx.x * 256 + threadIdx.x) * 4;
  if (i + 4 <= n) {
    const float4 v = *(const float4*)(in + i);
    ushort4 o;
    o.x = f32_to_bf16(v.x); o.y = f32_to_bf16(v.y);
    o.z = f32_to_bf16(v.z); o.w = f32_to_bf16(v.w);
    *(ushort4*)(out + i) = o;
  }
}

// in: [E,K,N] f32  ->  out: [E,N,K] bf16   (grid: N/32, K/32, E; 256 threads)
__global__ __launch_bounds__(256) void transpose_cvt_kernel(
    const float* __restrict__ in, uint16_t* __restrict__ out, int K, int N) {
  __shared__ float tile[32][33];
  const int e = blockIdx.z, k0 = blockIdx.y * 32, n0 = blockIdx.x * 32;
  const int tx = threadIdx.x & 31, ty = threadIdx.x >> 5;  // 32 x 8
  const float* ine  = in  + (size_t)e * K * N;
  uint16_t*    oute = out + (size_t)e * N * K;
  #pragma unroll
  for (int i = 0; i < 4; ++i)
    tile[ty + i * 8][tx] = ine[(size_t)(k0 + ty + i * 8) * N + (n0 + tx)];
  __syncthreads();
  #pragma unroll
  for (int i = 0; i < 4; ++i)
    oute[(size_t)(n0 + ty + i * 8) * K + (k0 + tx)] = f32_to_bf16(tile[tx][ty + i * 8]);
}

extern "C" void kernel_launch(void* const* d_in, const int* in_sizes, int n_in,
                              void* d_out, int out_size, void* d_ws, size_t ws_size,
                              hipStream_t stream) {
  constexpr int E = 8, T = 4096, H = 1024, I = 512;
  const float* x  = (const float*)d_in[0];
  const float* w1 = (const float*)d_in[1];
  const float* w2 = (const float*)d_in[2];
  const float* w3 = (const float*)d_in[3];
  const float* w4 = (const float*)d_in[4];

  uint16_t* xb  = (uint16_t*)d_ws;               // E*T*H bf16 (also reused as h2)
  uint16_t* w1t = xb  + (size_t)E * T * H;       // E*I*H
  uint16_t* w2t = w1t + (size_t)E * H * I;
  uint16_t* w3t = w2t + (size_t)E * H * I;
  uint16_t* w4t = w3t + (size_t)E * H * I;
  uint16_t* h1b = w4t + (size_t)E * H * I;       // E*T*I (also reused as h3)
  uint16_t* h2b = xb;                            // alias: xb dead after GEMM 1
  uint16_t* h3b = h1b;                           // alias: h1 dead after GEMM 2

  const size_t need =
      ((size_t)E * T * H + 4 * (size_t)E * H * I + (size_t)E * T * I) * sizeof(uint16_t);
  if (ws_size < need) return;

  { // x -> bf16
    const size_t n = (size_t)E * T * H;
    cvt_f32_to_bf16_kernel<<<dim3((unsigned)(n / 4 / 256)), 256, 0, stream>>>(x, xb, n);
  }
  // weights [K,N] -> bf16 [N,K] so GEMM B-fragments are K-contiguous
  transpose_cvt_kernel<<<dim3(I / 32, H / 32, E), 256, 0, stream>>>(w1, w1t, H, I);
  transpose_cvt_kernel<<<dim3(H / 32, I / 32, E), 256, 0, stream>>>(w2, w2t, I, H);
  transpose_cvt_kernel<<<dim3(I / 32, H / 32, E), 256, 0, stream>>>(w3, w3t, H, I);
  transpose_cvt_kernel<<<dim3(H / 32, I / 32, E), 256, 0, stream>>>(w4, w4t, I, H);

  gemm_bf16_wmma<true ><<<dim3(I / 128, T / 128, E), 256, 0, stream>>>(xb,  w1t, h1b, T, I, H);
  gemm_bf16_wmma<true ><<<dim3(H / 128, T / 128, E), 256, 0, stream>>>(h1b, w2t, h2b, T, H, I);
  gemm_bf16_wmma<true ><<<dim3(I / 128, T / 128, E), 256, 0, stream>>>(h2b, w3t, h3b, T, I, H);
  gemm_bf16_wmma<false><<<dim3(H / 128, T / 128, E), 256, 0, stream>>>(h3b, w4t, d_out, T, H, I);
}
